// LearnableEmbedding_38766374813998
// MI455X (gfx1250) — matrix operand order repaired and checked by reading, not compiled, not run on hardware
//
#include <hip/hip_runtime.h>
#include <hip/hip_bf16.h>
#include <math.h>

typedef __bf16 bf16;
typedef __attribute__((ext_vector_type(16))) __bf16 v16bf;
typedef __attribute__((ext_vector_type(8)))  float  v8f;
typedef __attribute__((ext_vector_type(4)))  unsigned int v4u;

namespace {
constexpr int   kT = 20, kB = 1024, kIn = 13, kOut = 9, kIters = 20;
constexpr float kLR = 0.1f, kB1 = 0.9f, kB2 = 0.999f, kEps = 1e-8f, kL2 = 1e-4f;
constexpr int   kBlocks = 16, kWaves = 4;

// -------- workspace layout (d_ws) --------
constexpr int SUM_F32   = 16;       // after 64B barrier area: 401 slots * 32 f32
constexpr int FRAG_BYTE = 65536;    // bf16 fragment region starts here
// fragment element offsets (bf16 elements, each frag = 512 elems = 1KB)
constexpr int F_hW1 = 0,      F_hW2 = 4096,   F_hW3 = 36864,  F_hW4 = 102400,
              F_WihT = 110592,F_WhhT = 112640,F_Wd = 113664,  F_WdT = 114176,
              F_WihS = 115200,F_hW4T = 116224,F_hW3T = 124416,F_hW2T = 189952,
              F_hW1T = 222720,F_BIAS = 226816;
constexpr int FB_b1 = 0, FB_b2 = 128, FB_b3 = 384, FB_b4 = 640, FB_bz = 672, FB_bd = 704;

// -------- per-wave LDS layout (bf16 elements) --------
constexpr int W_S = 0, W_A1 = 512, W_A2 = 2560, W_A3 = 6656, W_W = 10752,
              W_CAT = 11264, W_H = 12288, W_G1 = 12800, W_G2 = 16896,
              WAVE_ELEMS = 20992;
constexpr int SMEM_BYTES = kWaves * WAVE_ELEMS * 2 + kWaves * 256 * 4; // 172032
} // namespace

union FragU { v4u u[2]; v16bf b; };

// A-fragment (16x32 bf16) from row-major LDS tile, per CDNA5 ISA A layout:
// lanes 0-15: K 0..7 & 16..23 ; lanes 16-31: K 8..15 & 24..31
__device__ __forceinline__ v16bf ldsA(const bf16* A, int ldk, int kt, int lane) {
  const int m = lane & 15, half = lane >> 4;
  const bf16* p = A + m * ldk + kt * 32 + half * 8;
  FragU f;
  f.u[0] = *reinterpret_cast<const v4u*>(p);        // ds_load_b128
  f.u[1] = *reinterpret_cast<const v4u*>(p + 16);   // ds_load_b128
  return f.b;
}

// B-fragment (32x16 bf16) from pre-swizzled global (L2-resident) weight array.
__device__ __forceinline__ v16bf ldgB(const bf16* Wf, int NT, int kt, int nt, int lane) {
  const bf16* p = Wf + (((size_t)(kt * NT + nt)) << 9) + lane * 16;
  __builtin_prefetch(p + ((size_t)NT << 9), 0, 3);  // next k-tile -> global_prefetch_b8
  FragU f;
  f.u[0] = *reinterpret_cast<const v4u*>(p);        // global_load_b128
  f.u[1] = *reinterpret_cast<const v4u*>(p + 8);    // global_load_b128
  return f.b;
}

__device__ __forceinline__ v8f wmma_bf16(v16bf a, v16bf b, v8f c) {
  return __builtin_amdgcn_wmma_f32_16x16x32_bf16(false, a, false, b, (short)0, c,
                                                 false, false);
}

// D[16xN] = A[16xK] @ W + bias ; optional ReLU, optional relu-mask (bwd), f32 out.
// Processes TWO n-tiles per pass with dual accumulators sharing one A-fragment:
// halves A ds-traffic and issues independent back-to-back WMMAs so the bf16
// WMMA->VALU hazard (4 coexec NOPs, ISA 7.12.1) is amortized/hidden.
// All call sites have even NT.
__device__ __forceinline__ void wave_gemm(const bf16* A, int K, const bf16* Wf, int N,
                                          const bf16* bias, bf16* outb, float* outf,
                                          const bf16* mask, bool relu, int lane) {
  const int KT = K >> 5, NT = N >> 4;
  const int n = lane & 15, half = lane >> 4;
  for (int nt = 0; nt < NT; nt += 2) {
    const float bv0 = bias ? (float)bias[nt * 16 + n] : 0.f;
    const float bv1 = bias ? (float)bias[nt * 16 + 16 + n] : 0.f;
    v8f acc0, acc1;
#pragma unroll
    for (int r = 0; r < 8; ++r) { acc0[r] = bv0; acc1[r] = bv1; }
    for (int kt = 0; kt < KT; ++kt) {
      const v16bf a = ldsA(A, K, kt, lane);
      const v16bf b0 = ldgB(Wf, NT, kt, nt, lane);
      const v16bf b1 = ldgB(Wf, NT, kt, nt + 1, lane);
      acc0 = wmma_bf16(a, b0, acc0);   // independent pair: pipeline stays fed
      acc1 = wmma_bf16(a, b1, acc1);
    }
#pragma unroll
    for (int r = 0; r < 8; ++r) {
      const int m = r + half * 8;
      const int o0 = m * N + nt * 16 + n;
      const int o1 = o0 + 16;
      float v0 = acc0[r], v1 = acc1[r];
      if (relu) { v0 = fmaxf(v0, 0.f); v1 = fmaxf(v1, 0.f); }
      if (mask) {
        v0 = ((float)mask[o0] > 0.f) ? v0 : 0.f;
        v1 = ((float)mask[o1] > 0.f) ? v1 : 0.f;
      }
      if (outf) { outf[o0] = v0; outf[o1] = v1; }
      else      { outb[o0] = (bf16)v0; outb[o1] = (bf16)v1; }
    }
  }
}

// tanh RNN cell: z = cat@Wih^T + h@Whh^T + (bih+bhh); h_new stored bf16 to aH,
// f32 kept in hn[2] (C-layout tiles) for the backward. Both n-tiles use dual
// accumulators over shared A-fragments.
__device__ __forceinline__ void rnn_forward(const bf16* aCat, bf16* aH,
                                            const bf16* FWihT, const bf16* FWhhT,
                                            const bf16* bz, int lane, v8f* hn) {
  const int n = lane & 15, half = lane >> 4;
  const float bv0 = (float)bz[n];
  const float bv1 = (float)bz[16 + n];
  v8f acc0, acc1;
#pragma unroll
  for (int r = 0; r < 8; ++r) { acc0[r] = bv0; acc1[r] = bv1; }
#pragma unroll
  for (int kt = 0; kt < 2; ++kt) {
    const v16bf a = ldsA(aCat, 64, kt, lane);
    acc0 = wmma_bf16(a, ldgB(FWihT, 2, kt, 0, lane), acc0);
    acc1 = wmma_bf16(a, ldgB(FWihT, 2, kt, 1, lane), acc1);
  }
  {
    const v16bf a = ldsA(aH, 32, 0, lane);
    acc0 = wmma_bf16(a, ldgB(FWhhT, 2, 0, 0, lane), acc0);
    acc1 = wmma_bf16(a, ldgB(FWhhT, 2, 0, 1, lane), acc1);
  }
#pragma unroll
  for (int r = 0; r < 8; ++r) { acc0[r] = tanhf(acc0[r]); acc1[r] = tanhf(acc1[r]); }
  hn[0] = acc0; hn[1] = acc1;
#pragma unroll
  for (int r = 0; r < 8; ++r) {
    aH[(r + half * 8) * 32 + n]      = (bf16)acc0[r];
    aH[(r + half * 8) * 32 + 16 + n] = (bf16)acc1[r];
  }
}

__device__ __forceinline__ void grid_barrier(unsigned* bar) {
  __syncthreads();
  if (threadIdx.x == 0) {
    __threadfence();  // make this block's atomics visible device-wide
    const unsigned g =
        __hip_atomic_load(&bar[1], __ATOMIC_ACQUIRE, __HIP_MEMORY_SCOPE_AGENT);
    if (__hip_atomic_fetch_add(&bar[0], 1u, __ATOMIC_ACQ_REL,
                               __HIP_MEMORY_SCOPE_AGENT) == (unsigned)(kBlocks - 1)) {
      __hip_atomic_store(&bar[0], 0u, __ATOMIC_RELAXED, __HIP_MEMORY_SCOPE_AGENT);
      __hip_atomic_fetch_add(&bar[1], 1u, __ATOMIC_RELEASE, __HIP_MEMORY_SCOPE_AGENT);
    } else {
      while (__hip_atomic_load(&bar[1], __ATOMIC_ACQUIRE,
                               __HIP_MEMORY_SCOPE_AGENT) == g)
        __builtin_amdgcn_s_sleep(2);
    }
  }
  __syncthreads();
}

// ---------------- init: swizzle fp32 weights -> bf16 WMMA B-fragments ----------------
__device__ void fillB(bf16* dst, const float* src, int KT, int NT, int srcR, int srcC,
                      bool transpose, int colOff, int tid, int nth) {
  const int total = KT * NT * 512;
  for (int i = tid; i < total; i += nth) {
    const int f = i >> 9, r = i & 511;
    const int lane = r >> 4, j = r & 15;
    const int nt = f % NT, kt = f / NT;
    const int k = kt * 32 + ((j >> 3) << 4) + ((lane >> 4) << 3) + (j & 7);   // lanes0-15: K 0..7 & 16..23, 16-31: 8..15 & 24..31
    const int nn = nt * 16 + (lane & 15);
    float v = 0.f;
    if (!transpose) {
      const int sn = nn + colOff;
      if (k < srcR && sn < srcC) v = src[k * srcC + sn];
    } else {
      if (nn < srcR && k < srcC) v = src[nn * srcC + k];
    }
    dst[i] = (bf16)v;
  }
}

__global__ void init_kernel(const float* hW1, const float* hb1, const float* hW2,
                            const float* hb2, const float* hW3, const float* hb3,
                            const float* hW4, const float* hb4, const float* Wih,
                            const float* Whh, const float* bih, const float* bhh,
                            const float* Wd, const float* bd, float* ws, float* out) {
  const int tid = blockIdx.x * blockDim.x + threadIdx.x;
  const int nth = gridDim.x * blockDim.x;
  bf16* F = reinterpret_cast<bf16*>(reinterpret_cast<char*>(ws) + FRAG_BYTE);
  // forward fragments (zero-padded so padded A columns are annihilated)
  fillB(F + F_hW1,  hW1, 1, 8, 32, 128, false, 0, tid, nth);
  fillB(F + F_hW2,  hW2, 4, 16, 128, 256, false, 0, tid, nth);
  fillB(F + F_hW3,  hW3, 8, 16, 256, 256, false, 0, tid, nth);
  fillB(F + F_hW4,  hW4, 8, 2, 256, 32, false, 0, tid, nth);
  fillB(F + F_WihT, Wih, 2, 2, 32, 45, true, 0, tid, nth);    // B[k,n]=Wih[n][k], K pad 64
  fillB(F + F_WhhT, Whh, 1, 2, 32, 32, true, 0, tid, nth);
  fillB(F + F_Wd,   Wd,  1, 1, 32, 9, false, 0, tid, nth);    // N pad 16
  // backward (transposed) fragments
  fillB(F + F_WdT,  Wd,  1, 2, 32, 9, true, 0, tid, nth);     // K pad 32, zero rows k>=9
  fillB(F + F_WihS, Wih, 1, 2, 32, 45, false, 13, tid, nth);  // Wih[:,13:45]
  fillB(F + F_hW4T, hW4, 1, 16, 256, 32, true, 0, tid, nth);
  fillB(F + F_hW3T, hW3, 8, 16, 256, 256, true, 0, tid, nth);
  fillB(F + F_hW2T, hW2, 8, 8, 128, 256, true, 0, tid, nth);
  fillB(F + F_hW1T, hW1, 4, 2, 32, 128, true, 0, tid, nth);
  // biases (bf16)
  bf16* FB = F + F_BIAS;
  for (int i = tid; i < 128; i += nth) FB[FB_b1 + i] = (bf16)hb1[i];
  for (int i = tid; i < 256; i += nth) FB[FB_b2 + i] = (bf16)hb2[i];
  for (int i = tid; i < 256; i += nth) FB[FB_b3 + i] = (bf16)hb3[i];
  for (int i = tid; i < 32;  i += nth) FB[FB_b4 + i] = (bf16)hb4[i];
  for (int i = tid; i < 32;  i += nth) FB[FB_bz + i] = (bf16)(bih[i] + bhh[i]);
  for (int i = tid; i < 16;  i += nth) FB[FB_bd + i] = (bf16)((i < 9) ? bd[i] : 0.f);
  // zero barrier state + 401 reduction slots + output
  unsigned* z = reinterpret_cast<unsigned*>(ws);
  const int zcnt = SUM_F32 + 401 * 32;
  for (int i = tid; i < zcnt; i += nth) z[i] = 0u;
  for (int i = tid; i < 33; i += nth) out[i] = 0.f;
}

// ---------------- persistent main kernel: 16 blocks x 4 waves, 16 rows/wave ----------------
__global__ void __launch_bounds__(128, 1)
meta_kernel(const float* __restrict__ X, const float* __restrict__ Y,
            float* __restrict__ ws, float* __restrict__ out) {
  extern __shared__ char smem[];
  const int tid = threadIdx.x, lane = tid & 31, w = tid >> 5;
  const int rowBase = (blockIdx.x * kWaves + w) * 16;
  bf16* wb = reinterpret_cast<bf16*>(smem) + w * WAVE_ELEMS;
  bf16 *aS = wb + W_S, *aA1 = wb + W_A1, *aA2 = wb + W_A2, *aA3 = wb + W_A3,
       *aW = wb + W_W, *aCat = wb + W_CAT, *aH = wb + W_H,
       *g1 = wb + W_G1, *g2 = wb + W_G2;
  float* yb = reinterpret_cast<float*>(smem + kWaves * WAVE_ELEMS * 2) + w * 256;
  const bf16* F = reinterpret_cast<const bf16*>(reinterpret_cast<const char*>(ws) + FRAG_BYTE);
  const bf16* FB = F + F_BIAS;
  float* sums = ws + SUM_F32;
  unsigned* bar = reinterpret_cast<unsigned*>(ws);
  const int n = lane & 15, half = lane >> 4;
  const int mrow = lane & 15, c0 = half * 16;   // this lane owns story row mrow, cols c0..c0+15

  for (int i = lane; i < WAVE_ELEMS; i += 32) wb[i] = (bf16)0.f;  // story0=0, pads=0
  for (int i = lane; i < 256; i += 32) yb[i] = 0.f;

  float s[16], ma[16], va[16];
#pragma unroll
  for (int j = 0; j < 16; ++j) s[j] = 0.f;

  for (int t = 0; t < kT; ++t) {
#pragma unroll
    for (int j = 0; j < 16; ++j) { ma[j] = 0.f; va[j] = 0.f; }   // fresh Adam
    float b1p = 1.f, b2p = 1.f;
    for (int i = lane; i < 512; i += 32) aH[i] = (bf16)0.f;      // h0 = 0
    for (int i = lane; i < 256; i += 32) yb[i] = 0.f;
    for (int i = lane; i < 16 * kIn; i += 32) {                  // x_t -> cat[:, :13]
      const int m = i / kIn, c = i % kIn;
      aCat[m * 64 + c] = (bf16)X[((size_t)t * kB + rowBase + m) * kIn + c];
    }
    for (int i = lane; i < 16 * kOut; i += 32) {                 // y_t -> LDS (f32)
      const int m = i / kOut, c = i % kOut;
      yb[m * 16 + c] = Y[((size_t)t * kB + rowBase + m) * kOut + c];
    }

    for (int it = 0; it < kIters; ++it) {
      const int idx = t * kIters + it;
      {  // story column partial sums (for the var-term mean): one atomic per lane
        float cs = 0.f;
        for (int r = 0; r < 16; ++r) cs += (float)aS[r * 32 + lane];
        atomicAdd(&sums[idx * 32 + lane], cs);
      }
      // ---- hypernet forward ----
      wave_gemm(aS, 32, F + F_hW1, 128, FB + FB_b1, aA1, nullptr, nullptr, true, lane);
      wave_gemm(aA1, 128, F + F_hW2, 256, FB + FB_b2, aA2, nullptr, nullptr, true, lane);
      wave_gemm(aA2, 256, F + F_hW3, 256, FB + FB_b3, aA3, nullptr, nullptr, true, lane);
      wave_gemm(aA3, 256, F + F_hW4, 32, FB + FB_b4, aW, nullptr, nullptr, false, lane);
      for (int i = lane; i < 512; i += 32) {                     // w -> cat[:, 13:45]
        const int m = i >> 5, c = i & 31;
        aCat[m * 64 + 13 + c] = aW[i];
      }
      // ---- RNN cell (uses previous hidden, stores new hidden) ----
      v8f hn[2];
      rnn_forward(aCat, aH, F + F_WihT, F + F_WhhT, FB + FB_bz, lane, hn);
      // ---- decoder + dpred = relu'(pred) * 2(pred-y)/B ----
      {
        float bv = (float)FB[FB_bd + n];
        v8f acc;
#pragma unroll
        for (int r = 0; r < 8; ++r) acc[r] = bv;
        acc = wmma_bf16(ldsA(aH, 32, 0, lane), ldgB(F + F_Wd, 1, 0, 0, lane), acc);
#pragma unroll
        for (int r = 0; r < 8; ++r) {
          const int m = r + half * 8;
          const float pv = acc[r];
          const float dv = (pv > 0.f) ? (2.0f / (float)kB) * (pv - yb[m * 16 + n]) : 0.f;
          g1[m * 32 + n] = (bf16)dv;
        }
      }
      // ---- dh = dpred@Wd^T ; dz = dh*(1-h^2) : dual accumulators, shared A ----
      {
        v8f acc0, acc1;
#pragma unroll
        for (int r = 0; r < 8; ++r) { acc0[r] = 0.f; acc1[r] = 0.f; }
        const v16bf a = ldsA(g1, 32, 0, lane);
        acc0 = wmma_bf16(a, ldgB(F + F_WdT, 2, 0, 0, lane), acc0);
        acc1 = wmma_bf16(a, ldgB(F + F_WdT, 2, 0, 1, lane), acc1);
#pragma unroll
        for (int r = 0; r < 8; ++r) {
          const float h0 = hn[0][r], h1 = hn[1][r];
          g2[(r + half * 8) * 32 + n]      = (bf16)(acc0[r] * (1.f - h0 * h0));
          g2[(r + half * 8) * 32 + 16 + n] = (bf16)(acc1[r] * (1.f - h1 * h1));
        }
      }
      // ---- backward through hypernet (relu masks from forward acts) ----
      wave_gemm(g2, 32, F + F_WihS, 32, nullptr, g1, nullptr, nullptr, false, lane);  // dw
      wave_gemm(g1, 32, F + F_hW4T, 256, nullptr, g2, nullptr, aA3, false, lane);     // da3
      wave_gemm(g2, 256, F + F_hW3T, 256, nullptr, g1, nullptr, aA2, false, lane);    // da2
      wave_gemm(g1, 256, F + F_hW2T, 128, nullptr, g2, nullptr, aA1, false, lane);    // da1
      wave_gemm(g2, 128, F + F_hW1T, 32, nullptr, nullptr,
                reinterpret_cast<float*>(g1), nullptr, false, lane);                  // dstory f32

      grid_barrier(bar);  // all column sums for `idx` now visible

      // ---- Adam update (story/m/v live in VGPRs) ----
      b1p *= kB1; b2p *= kB2;
      const float* gsf = reinterpret_cast<const float*>(g1);
#pragma unroll
      for (int j = 0; j < 16; ++j) {
        const int c = c0 + j;
        float g = gsf[mrow * 32 + c];
        const float mean = sums[idx * 32 + c] * (1.f / (float)kB);
        g += 2.f * (s[j] - mean) / (float)(kB - 1) + (2.f * kL2 / (float)kB) * s[j];
        ma[j] = kB1 * ma[j] + (1.f - kB1) * g;
        va[j] = kB2 * va[j] + (1.f - kB2) * g * g;
        const float mh = ma[j] / (1.f - b1p);
        const float vh = va[j] / (1.f - b2p);
        s[j] -= kLR * mh / (sqrtf(vh) + kEps);
        aS[mrow * 32 + c] = (bf16)s[j];
      }
    }
  }

  // ---- cluster_centers = mean(story, axis=0) ----
  {
    float cs = 0.f;
    for (int r = 0; r < 16; ++r) cs += (float)aS[r * 32 + lane];
    atomicAdd(&sums[kT * kIters * 32 + lane], cs);
  }
  grid_barrier(bar);
  if (blockIdx.x == 0 && tid < 32)
    out[1 + tid] = sums[kT * kIters * 32 + tid] * (1.f / (float)kB);

  // ---- final evaluation pass ----
  wave_gemm(aS, 32, F + F_hW1, 128, FB + FB_b1, aA1, nullptr, nullptr, true, lane);
  wave_gemm(aA1, 128, F + F_hW2, 256, FB + FB_b2, aA2, nullptr, nullptr, true, lane);
  wave_gemm(aA2, 256, F + F_hW3, 256, FB + FB_b3, aA3, nullptr, nullptr, true, lane);
  wave_gemm(aA3, 256, F + F_hW4, 32, FB + FB_b4, aW, nullptr, nullptr, false, lane);
  for (int i = lane; i < 512; i += 32) {
    const int m = i >> 5, c = i & 31;
    aCat[m * 64 + 13 + c] = aW[i];
  }
  for (int i = lane; i < 512; i += 32) aH[i] = (bf16)0.f;
  float errAcc = 0.f;
  for (int t = 0; t < kT; ++t) {
    for (int i = lane; i < 256; i += 32) yb[i] = 0.f;
    for (int i = lane; i < 16 * kIn; i += 32) {
      const int m = i / kIn, c = i % kIn;
      aCat[m * 64 + c] = (bf16)X[((size_t)t * kB + rowBase + m) * kIn + c];
    }
    for (int i = lane; i < 16 * kOut; i += 32) {
      const int m = i / kOut, c = i % kOut;
      yb[m * 16 + c] = Y[((size_t)t * kB + rowBase + m) * kOut + c];
    }
    v8f hn[2];
    rnn_forward(aCat, aH, F + F_WihT, F + F_WhhT, FB + FB_bz, lane, hn);
    {
      float bv = (float)FB[FB_bd + n];
      v8f acc;
#pragma unroll
      for (int r = 0; r < 8; ++r) acc[r] = bv;
      acc = wmma_bf16(ldsA(aH, 32, 0, lane), ldgB(F + F_Wd, 1, 0, 0, lane), acc);
#pragma unroll
      for (int r = 0; r < 8; ++r) {
        const int m = r + half * 8;
        const float pv = fmaxf(acc[r], 0.f);
        const float d = pv - yb[m * 16 + n];  // cols >=9: pred=0, y=0 -> no contribution
        errAcc += d * d;
      }
    }
  }
  atomicAdd(&out[0], errAcc * (1.f / (float)kB));
}

extern "C" void kernel_launch(void* const* d_in, const int* in_sizes, int n_in,
                              void* d_out, int out_size, void* d_ws, size_t ws_size,
                              hipStream_t stream) {
  (void)in_sizes; (void)n_in; (void)out_size; (void)ws_size;
  const float* X = (const float*)d_in[0];
  const float* Y = (const float*)d_in[1];
  init_kernel<<<dim3(64), dim3(256), 0, stream>>>(
      (const float*)d_in[2], (const float*)d_in[3], (const float*)d_in[4],
      (const float*)d_in[5], (const float*)d_in[6], (const float*)d_in[7],
      (const float*)d_in[8], (const float*)d_in[9], (const float*)d_in[10],
      (const float*)d_in[11], (const float*)d_in[12], (const float*)d_in[13],
      (const float*)d_in[14], (const float*)d_in[15], (float*)d_ws, (float*)d_out);
  meta_kernel<<<dim3(kBlocks), dim3(kWaves * 32), SMEM_BYTES, stream>>>(
      X, Y, (float*)d_ws, (float*)d_out);
}
